// OuroborosMoELayer_83571473645700
// MI455X (gfx1250) — matrix operand order repaired and checked by reading, not compiled, not run on hardware
//
#include <hip/hip_runtime.h>
#include <hip/hip_bf16.h>
#include <math.h>

#define B_  8
#define T_  2048
#define D_  1024
#define FF_ 4096
#define E_  8
#define K_  2

typedef __bf16 bf16_t;
typedef __attribute__((ext_vector_type(16))) __bf16 bf16x16;
typedef __attribute__((ext_vector_type(8)))  __bf16 bf16x8;
typedef __attribute__((ext_vector_type(8)))  float  f32x8;
typedef __attribute__((ext_vector_type(4)))  unsigned int u32x4;
typedef __attribute__((ext_vector_type(8)))  int i32x8;
typedef __attribute__((ext_vector_type(4)))  int i32x4;

// x tile LDS layout produced by the TDM with pad_enable:
//   row stride = 2048B data + 2x16B pads = 2080B (= 1040 bf16 elements)
//   a 16B pad is inserted after each 256 DWORDs (512 bf16 elements)
#define XLD 1040   // elements per row incl. padding
#define HLD 136    // h chunk: 128 + 8 pad

// ---------------------------------------------------------------------------
// WMMA helpers
// ---------------------------------------------------------------------------
__device__ __forceinline__ f32x8 wmma_bf16(bf16x16 a, bf16x16 b, f32x8 c) {
  // (neg_a, A, neg_b, B, c_mod, C, reuse_a, reuse_b)
  return __builtin_amdgcn_wmma_f32_16x16x32_bf16(false, a, false, b,
                                                 (short)0, c, false, false);
}

// A-operand fragment (16x32 bf16), contiguous K order:
//   lanes 0-15 : row M=lane,    elems 0-15 = K[dk..dk+15]
//   lanes 16-31: row M=lane-16, elems 0-15 = K[dk+16..dk+31]
__device__ __forceinline__ bf16x16 load_a_frag(const bf16_t* p) {
  bf16x8 lo = *(const bf16x8*)p;
  bf16x8 hi = *(const bf16x8*)(p + 8);
  bf16x16 a;
#pragma unroll
  for (int i = 0; i < 8; ++i) { a[i] = lo[i]; a[i + 8] = hi[i]; }
  return a;
}

// ---------------------------------------------------------------------------
// Routing: pooled mean -> logits -> top-2 -> softmax weights
// ---------------------------------------------------------------------------
__global__ __launch_bounds__(256) void routing_kernel(
    const float* __restrict__ x, const float* __restrict__ Wr,
    int* __restrict__ ridx, float* __restrict__ rw)
{
  __shared__ float pooled[D_];
  __shared__ float logits[E_];
  const int b   = blockIdx.x;
  const int tid = threadIdx.x;

  float a0 = 0.f, a1 = 0.f, a2 = 0.f, a3 = 0.f;
  const float* xb = x + (size_t)b * T_ * D_;
  for (int t = 0; t < T_; ++t) {
    const float4 v = *(const float4*)(xb + (size_t)t * D_ + tid * 4);
    a0 += v.x; a1 += v.y; a2 += v.z; a3 += v.w;
  }
  const float inv = 1.0f / (float)T_;
  pooled[tid * 4 + 0] = a0 * inv;
  pooled[tid * 4 + 1] = a1 * inv;
  pooled[tid * 4 + 2] = a2 * inv;
  pooled[tid * 4 + 3] = a3 * inv;
  __syncthreads();

  if (tid < E_) {
    float s = 0.0f;
    for (int d = 0; d < D_; ++d) s += pooled[d] * Wr[d * E_ + tid];
    logits[tid] = s;
  }
  __syncthreads();

  if (tid == 0) {
    int i0 = 0;
    for (int e = 1; e < E_; ++e) if (logits[e] > logits[i0]) i0 = e;
    int i1 = (i0 == 0) ? 1 : 0;
    for (int e = 0; e < E_; ++e)
      if (e != i0 && logits[e] > logits[i1]) i1 = e;
    const float l0 = logits[i0], l1 = logits[i1];
    const float e1f = expf(l1 - l0);           // l0 is the max
    const float s   = 1.0f + e1f;
    ridx[b * 2 + 0] = i0;  ridx[b * 2 + 1] = i1;
    rw[b * 2 + 0]   = 1.0f / s;
    rw[b * 2 + 1]   = e1f / s;
  }
}

// ---------------------------------------------------------------------------
// fp32 -> bf16 copy
// ---------------------------------------------------------------------------
__global__ void cvt_bf16_kernel(const float* __restrict__ in,
                                bf16_t* __restrict__ outp, long long n)
{
  long long i = (long long)blockIdx.x * blockDim.x + threadIdx.x;
  const long long stride = (long long)gridDim.x * blockDim.x;
  for (; i < n; i += stride) outp[i] = (bf16_t)in[i];
}

// fp32 [e][r][c] -> bf16 [e][c][r]  (per-expert transpose + downconvert)
__global__ void transpose_cvt_kernel(const float* __restrict__ in,
                                     bf16_t* __restrict__ outp,
                                     int R, int C, long long n)
{
  long long i = (long long)blockIdx.x * blockDim.x + threadIdx.x;
  const long long stride = (long long)gridDim.x * blockDim.x;
  const long long rc = (long long)R * C;
  for (; i < n; i += stride) {
    const long long e   = i / rc;
    const long long rem = i - e * rc;
    const int r = (int)(rem / C);
    const int c = (int)(rem - (long long)r * C);
    outp[e * rc + (long long)c * R + r] = (bf16_t)in[i];
  }
}

// ---------------------------------------------------------------------------
// Fused MoE FFN: per block = 16 token rows of one sample.
//   x tile staged LDS via the Tensor Data Mover (async, TENSORcnt-fenced),
//   for k in {0,1}: per 128-wide f-chunk: GEMM1(x@W1)+b1 -> GELU*w_k -> LDS
//                                         GEMM2(h@W2) accumulated in regs
//   out = acc + sum_k w_k*b2[e_k]
// ---------------------------------------------------------------------------
__global__ __launch_bounds__(256) void moe_ffn_kernel(
    const bf16_t* __restrict__ xbf,  const bf16_t* __restrict__ W1t,
    const bf16_t* __restrict__ W2t,  const float* __restrict__ b1,
    const float* __restrict__ b2,    const int* __restrict__ ridx,
    const float* __restrict__ rw,    float* __restrict__ out)
{
  __shared__ __align__(16) bf16_t xs[16 * XLD];   // x row tile (TDM-padded)
  __shared__ __align__(16) bf16_t hs[16 * HLD];   // h f-chunk

  const int b    = blockIdx.y;
  const int t0   = blockIdx.x * 16;
  const int tid  = threadIdx.x;
  const int wv   = tid >> 5;          // wave 0..7
  const int lane = tid & 31;
  const int half = lane >> 4;         // 0: lanes 0-15, 1: lanes 16-31
  const int l15  = lane & 15;

  // ----- stage x tile (16 x 1024 bf16) into LDS via the Tensor Data Mover --
  const unsigned long long gaddr =
      (unsigned long long)(uintptr_t)(xbf + (size_t)(b * T_ + t0) * D_);
  const unsigned lds_off = (unsigned)(uintptr_t)(&xs[0]);  // low 32b = LDS byte offset

#if __has_builtin(__builtin_amdgcn_tensor_load_to_lds)
  if (tid == 0) {
    // D# group 0: count=1 (valid), lds_addr, global_addr[56:0], type=2
    u32x4 g0;
    g0[0] = 1u;
    g0[1] = lds_off;
    g0[2] = (unsigned)(gaddr & 0xffffffffull);
    g0[3] = (unsigned)((gaddr >> 32) & 0x01ffffffull) | (2u << 30);
    // D# group 1: data_size=2B, pad_enable, pad_interval=256dw, pad_amount=4dw
    //             tensor_dim0=1024, tensor_dim1=2048, tile_dim0=1024,
    //             tile_dim1=16, tensor_dim0_stride=1024
    i32x8 g1;
    g1[0] = (1 << 16) | (1 << 20) | (7 << 22) | (3 << 25);
    g1[1] = (int)(1024u << 16);        // [63:48]  tensor_dim0 lo16
    g1[2] = (int)(2048u << 16);        // [95:80]  tensor_dim1 lo16
    g1[3] = (int)(1024u << 16);        // [127:112] tile_dim0
    g1[4] = 16;                        // [143:128] tile_dim1 (tile_dim2=0)
    g1[5] = 1024;                      // [191:160] tensor_dim0_stride lo32
    g1[6] = 0;
    g1[7] = 0;
    const i32x4 gz = (i32x4){0, 0, 0, 0};   // groups 2/3: 2D tensor, unused dims
#if defined(__clang_major__) && (__clang_major__ >= 23)
    const i32x8 gz8 = (i32x8){0, 0, 0, 0, 0, 0, 0, 0};
    __builtin_amdgcn_tensor_load_to_lds(g0, g1, gz, gz, gz8, 0);
#else
    __builtin_amdgcn_tensor_load_to_lds(g0, g1, gz, gz, 0);
#endif
#if __has_builtin(__builtin_amdgcn_s_wait_tensorcnt)
    __builtin_amdgcn_s_wait_tensorcnt(0);
#else
    asm volatile("s_wait_tensorcnt 0x0" ::: "memory");
#endif
  }
#else
  // Fallback: cooperative load into the same TDM-padded layout
  for (int i = tid; i < 16 * 64; i += 256) {
    const int r  = i >> 6;
    const int cc = i & 63;                      // 16-element chunk in row
    const int off = cc * 16 + ((cc >= 32) ? 8 : 0);
    *(uint4*)&xs[r * XLD + off] =
        *(const uint4*)&xbf[((size_t)(b * T_ + t0 + r)) * D_ + cc * 16];
  }
#endif
  __syncthreads();   // TDM data visible to all 8 waves

  f32x8 acc[8];
  float biasv[8];
#pragma unroll
  for (int ct = 0; ct < 8; ++ct) {
    acc[ct]   = (f32x8){0.f,0.f,0.f,0.f,0.f,0.f,0.f,0.f};
    biasv[ct] = 0.0f;
  }

  for (int k = 0; k < K_; ++k) {
    const int    e   = ridx[b * K_ + k];
    const float  wk  = rw[b * K_ + k];
    const bf16_t* W1e = W1t + (size_t)e * FF_ * D_;   // [f][d]
    const bf16_t* W2e = W2t + (size_t)e * D_ * FF_;   // [d][f]

#pragma unroll
    for (int ct = 0; ct < 8; ++ct)
      biasv[ct] += wk * b2[e * D_ + wv * 128 + ct * 16 + l15];

    for (int fb = 0; fb < FF_; fb += 128) {
      // ---------- GEMM1: wave wv owns h columns [fb+wv*16, +16) ----------
      const int fcol = fb + wv * 16 + l15;            // this lane's N column
      const bf16_t* bp = W1e + (size_t)fcol * D_;     // contiguous in K=d
      __builtin_prefetch((const void*)(bp + (size_t)128 * D_), 0, 0);

      f32x8 hacc = (f32x8){0.f,0.f,0.f,0.f,0.f,0.f,0.f,0.f};
#pragma unroll 4
      for (int dk = 0; dk < D_; dk += 32) {
        const int dkOff = dk + ((dk & 512) >> 6);     // +8 elems past mid-row pad
        bf16x16 a  = load_a_frag(&xs[l15 * XLD + dkOff + half * 16]);
        bf16x16 bb = *(const bf16x16*)(bp + dk + half * 16);
        hacc = wmma_bf16(a, bb, hacc);
      }

      // bias + exact-erf GELU, pre-scale by routing weight, -> LDS (bf16)
      const float b1v = b1[e * FF_ + fcol];
#pragma unroll
      for (int j = 0; j < 8; ++j) {
        float v = hacc[j] + b1v;
        v = 0.5f * v * (1.0f + erff(v * 0.70710678118654752f));
        hs[(half * 8 + j) * HLD + wv * 16 + l15] = (bf16_t)(wk * v);
      }
      __syncthreads();

      // ---------- GEMM2: wave wv owns D columns [wv*128, +128) ----------
#pragma unroll
      for (int ks = 0; ks < 4; ++ks) {
        bf16x16 a2 = load_a_frag(&hs[l15 * HLD + ks * 32 + half * 16]);
#pragma unroll
        for (int ct = 0; ct < 8; ++ct) {
          const bf16_t* b2p = W2e
              + (size_t)(wv * 128 + ct * 16 + l15) * FF_   // fixed N=d column
              + fb + ks * 32 + half * 16;                  // contiguous K=f
          bf16x16 bb2 = *(const bf16x16*)b2p;
          acc[ct] = wmma_bf16(a2, bb2, acc[ct]);
        }
      }
      __syncthreads();  // hs is rewritten next chunk
    }
  }

  // store fp32 output (C/D layout: VGPR j -> row j + 8*half, col = l15)
  float* ob = out + ((size_t)(b * T_) + t0) * D_;
#pragma unroll
  for (int ct = 0; ct < 8; ++ct) {
    const int col = wv * 128 + ct * 16 + l15;
#pragma unroll
    for (int j = 0; j < 8; ++j)
      ob[(size_t)(half * 8 + j) * D_ + col] = acc[ct][j] + biasv[ct];
  }
}

// ---------------------------------------------------------------------------
// Launcher
// ---------------------------------------------------------------------------
extern "C" void kernel_launch(void* const* d_in, const int* in_sizes, int n_in,
                              void* d_out, int out_size, void* d_ws, size_t ws_size,
                              hipStream_t stream)
{
  const float* x  = (const float*)d_in[0];   // (B,T,D)
  const float* Wr = (const float*)d_in[1];   // (D,E)
  const float* W1 = (const float*)d_in[2];   // (E,D,FF)
  const float* b1 = (const float*)d_in[3];   // (E,FF)
  const float* W2 = (const float*)d_in[4];   // (E,FF,D)
  const float* b2 = (const float*)d_in[5];   // (E,D)
  float* out = (float*)d_out;

  char* ws = (char*)d_ws;
  int*    ridx = (int*)ws;                         // 16 ints
  float*  rw   = (float*)(ws + 64);                // 16 floats
  bf16_t* xbf  = (bf16_t*)(ws + 256);
  const size_t xbytes = (size_t)B_ * T_ * D_ * sizeof(bf16_t);   // 32 MiB
  const size_t wbytes = (size_t)E_ * D_ * FF_ * sizeof(bf16_t);  // 64 MiB
  bf16_t* W1t = (bf16_t*)(ws + 256 + xbytes);            // [e][f][d] bf16
  bf16_t* W2t = (bf16_t*)(ws + 256 + xbytes + wbytes);   // [e][d][f] bf16

  routing_kernel<<<B_, 256, 0, stream>>>(x, Wr, ridx, rw);

  const long long nx = (long long)B_ * T_ * D_;
  cvt_bf16_kernel<<<4096, 256, 0, stream>>>(x, xbf, nx);

  const long long nw = (long long)E_ * D_ * FF_;
  transpose_cvt_kernel<<<8192, 256, 0, stream>>>(W1, W1t, D_, FF_, nw);
  transpose_cvt_kernel<<<8192, 256, 0, stream>>>(W2, W2t, FF_, D_, nw);

  dim3 grid(T_ / 16, B_);
  moe_ffn_kernel<<<grid, 256, 0, stream>>>(xbf, W1t, W2t, b1, b2, ridx, rw, out);
}
